// DDC_module_7421703487603
// MI455X (gfx1250) — matrix operand order repaired and checked by reading, not compiled, not run on hardware
//
#include <hip/hip_runtime.h>
#include <hip/hip_bf16.h>

typedef __attribute__((ext_vector_type(16))) _Float16 v16h;
typedef __attribute__((ext_vector_type(8)))  float    v8f;

#define B_   32
#define C_   512
#define CR_  256
#define H_   48
#define W_   48
#define KK   9          // 3x3 kernel taps
#define G_   2
#define EPSBN 1e-5f

// ---------------------------------------------------------------------------
// WMMA fragment gathers for V_WMMA_F32_16X16X32_F16 (ISA 7.12.2 layouts).
// A (16x32 f16): lane L -> row M = L&15; halves 0..7 = K (L<16?0:8)+i,
//                halves 8..15 = K 16+(L<16?0:8)+i.
// B (32x16 f16): lane L -> col N = L&15; halves 0..7 = K (L<16?0:8)+i,
//                halves 8..15 = K 16+(L<16?0:8)+i.
// C/D (16x16 f32): lane L -> col N = L&15; vgpr r -> row M = r + (L<16?0:8).
// ---------------------------------------------------------------------------
__device__ __forceinline__ v16h frag_a_f32(const float* __restrict__ A, int lda,
                                           int m0, int k0, int lane) {
  const float* row = A + (size_t)(m0 + (lane & 15)) * lda + k0 + ((lane >> 4) << 3);
  v16h a;
#pragma unroll
  for (int i = 0; i < 8; ++i) { a[i] = (_Float16)row[i]; a[i + 8] = (_Float16)row[16 + i]; }
  return a;
}
__device__ __forceinline__ v16h frag_a_f16(const _Float16* __restrict__ A, int lda,
                                           int m0, int k0, int lane) {
  const _Float16* row = A + (size_t)(m0 + (lane & 15)) * lda + k0 + ((lane >> 4) << 3);
  v16h a;
#pragma unroll
  for (int i = 0; i < 8; ++i) { a[i] = row[i]; a[i + 8] = row[16 + i]; }
  return a;
}
// B-fragment of W^T where W is (N x K) row-major (1x1-conv weight layout).
__device__ __forceinline__ v16h frag_bT_f32(const float* __restrict__ W, int ldw,
                                            int k0, int n0, int lane) {
  const float* row = W + (size_t)(n0 + (lane & 15)) * ldw + k0 + ((lane >> 4) << 3);
  v16h b;
#pragma unroll
  for (int i = 0; i < 8; ++i) { b[i] = (_Float16)row[i]; b[i + 8] = (_Float16)row[16 + i]; }
  return b;
}

// ---------------------------------------------------------------------------
// Kernel 1: 3x3 adaptive avg-pool. One block per (b,c); 9 waves; wave t
// reduces one 16x16 tile. Output written pre-transposed for GEMM:
// x1t[(b*9 + t) * C + c]. x stays L2-resident (151MB < 192MB L2) for pass 2.
// ---------------------------------------------------------------------------
__global__ void pool_kernel(const float* __restrict__ x, float* __restrict__ x1t) {
  int bc = blockIdx.x;                       // b*C + c
  int b = bc / C_, c = bc % C_;
  int wave = threadIdx.x >> 5;               // 0..8 = kh*3+kw
  int lane = threadIdx.x & 31;
  int tr = (wave / 3) * 16, tc = (wave % 3) * 16;
  const float* plane = x + (size_t)bc * (H_ * W_);
  float s = 0.f;
#pragma unroll
  for (int j = 0; j < 8; ++j) {
    int p = lane + j * 32;                   // 0..255 within tile
    s += plane[(tr + (p >> 4)) * W_ + (tc + (p & 15))];
  }
#pragma unroll
  for (int off = 16; off > 0; off >>= 1) s += __shfl_down(s, off);
  if (lane == 0) x1t[(size_t)(b * KK + wave) * C_ + c] = s * (1.f / 256.f);
}

// gap[b,c] = mean of the 9 equal-size tile means
__global__ void gap_kernel(const float* __restrict__ x1t, float* __restrict__ gapt) {
  int i = blockIdx.x * blockDim.x + threadIdx.x;
  if (i >= B_ * C_) return;
  int b = i / C_, c = i % C_;
  float s = 0.f;
#pragma unroll
  for (int t = 0; t < KK; ++t) s += x1t[(size_t)(b * KK + t) * C_ + c];
  gapt[i] = s * (1.f / 9.f);
}

// ---------------------------------------------------------------------------
// GEMM1 + BatchNorm + exact GELU; one wave per 16x16 output tile.
// A (MxK f32 row-major), W (NxK f32), out f16 (MxN row-major).
// ---------------------------------------------------------------------------
__global__ void gemm_bn_gelu_kernel(const float* __restrict__ A, const float* __restrict__ W,
                                    const float* __restrict__ bng, const float* __restrict__ bnb,
                                    const float* __restrict__ bnm, const float* __restrict__ bnv,
                                    _Float16* __restrict__ Hout, int N, int Kdim) {
  int m0 = blockIdx.x * 16, n0 = blockIdx.y * 16;
  int lane = threadIdx.x;
  v8f acc = {};
  for (int k0 = 0; k0 < Kdim; k0 += 32) {
    v16h a = frag_a_f32(A, Kdim, m0, k0, lane);
    v16h b = frag_bT_f32(W, Kdim, k0, n0, lane);
    acc = __builtin_amdgcn_wmma_f32_16x16x32_f16(false, a, false, b, (short)0, acc, false, false);
  }
  int n = n0 + (lane & 15);
  float inv = bng[n] * rsqrtf(bnv[n] + EPSBN);
  float mu = bnm[n], beta = bnb[n];
  int mbase = m0 + ((lane >> 4) << 3);
#pragma unroll
  for (int r = 0; r < 8; ++r) {
    float h = (acc[r] - mu) * inv + beta;
    h = 0.5f * h * (1.f + erff(h * 0.70710678118f));   // exact GELU
    Hout[(size_t)(mbase + r) * N + n] = (_Float16)h;
  }
}

// ---------------------------------------------------------------------------
// p-branch GEMM2 (288x256 . 256x512) fused with +bias, G=2 softmax against
// x1, and mixing of the two depthwise kernel banks -> per-sample 3x3 kernels.
// ---------------------------------------------------------------------------
__global__ void gemm_p2_fuse_kernel(const _Float16* __restrict__ Hh, const float* __restrict__ W2,
                                    const float* __restrict__ b2, const float* __restrict__ x1t,
                                    const float* __restrict__ dw_w, float* __restrict__ wout) {
  int m0 = blockIdx.x * 16, n0 = blockIdx.y * 16;
  int lane = threadIdx.x;
  v8f acc = {};
  for (int k0 = 0; k0 < CR_; k0 += 32) {
    v16h a = frag_a_f16(Hh, CR_, m0, k0, lane);
    v16h b = frag_bT_f32(W2, CR_, k0, n0, lane);
    acc = __builtin_amdgcn_wmma_f32_16x16x32_f16(false, a, false, b, (short)0, acc, false, false);
  }
  int n = n0 + (lane & 15);                 // channel c
  float bias2 = b2[n];
  int mbase = m0 + ((lane >> 4) << 3);
#pragma unroll
  for (int r = 0; r < 8; ++r) {
    int m = mbase + r;                      // b*9 + t
    float s  = acc[r] + bias2;              // scale1[b,c,t]  (group 0)
    float xv = x1t[(size_t)m * C_ + n];     // x1[b,c,t]      (group 1)
    float mx = fmaxf(s, xv);
    float e0 = expf(s - mx), e1 = expf(xv - mx);
    float p0 = e0 / (e0 + e1);
    int b = m / KK, t = m % KK;
    float wgt = p0 * dw_w[n * KK + t] + (1.f - p0) * dw_w[C_ * KK + n * KK + t];
    wout[((size_t)(b * C_ + n)) * KK + t] = wgt;
  }
}

// q-branch GEMM2 (32x256 . 256x1024) + bias -> logits
__global__ void gemm_bias_kernel(const _Float16* __restrict__ Hh, const float* __restrict__ W2,
                                 const float* __restrict__ b2, float* __restrict__ Sout,
                                 int N, int Kdim) {
  int m0 = blockIdx.x * 16, n0 = blockIdx.y * 16;
  int lane = threadIdx.x;
  v8f acc = {};
  for (int k0 = 0; k0 < Kdim; k0 += 32) {
    v16h a = frag_a_f16(Hh, Kdim, m0, k0, lane);
    v16h b = frag_bT_f32(W2, Kdim, k0, n0, lane);
    acc = __builtin_amdgcn_wmma_f32_16x16x32_f16(false, a, false, b, (short)0, acc, false, false);
  }
  int n = n0 + (lane & 15);
  float bias2 = b2[n];
  int mbase = m0 + ((lane >> 4) << 3);
#pragma unroll
  for (int r = 0; r < 8; ++r)
    Sout[(size_t)(mbase + r) * N + n] = acc[r] + bias2;
}

// softmax over G=2 logits, mix the two bias banks -> per-(b,c) bias
__global__ void bias_mix_kernel(const float* __restrict__ Sq, const float* __restrict__ dw_b,
                                float* __restrict__ bout) {
  int i = blockIdx.x * blockDim.x + threadIdx.x;
  if (i >= B_ * C_) return;
  int b = i / C_, c = i % C_;
  float s0 = Sq[(size_t)b * (G_ * C_) + c];
  float s1 = Sq[(size_t)b * (G_ * C_) + C_ + c];
  float mx = fmaxf(s0, s1);
  float e0 = expf(s0 - mx), e1 = expf(s1 - mx);
  float p0 = e0 / (e0 + e1);
  bout[i] = p0 * dw_b[c] + (1.f - p0) * dw_b[C_ + c];
}

// ---------------------------------------------------------------------------
// Kernel 4: depthwise 3x3 conv with per-(b,c) kernels + bias, zero padding.
// One block per plane. The 48x48 plane (9.2 KB) is staged into LDS via the
// CDNA5 async global->LDS DMA path (GLOBAL_LOAD_ASYNC_TO_LDS_B128, tracked by
// ASYNCcnt), which overlaps with the per-plane weight/bias loads; the LDS
// address for the async op is the low 32 bits of the flat shared pointer
// (aperture rule).
// ---------------------------------------------------------------------------
__global__ void dwconv_kernel(const float* __restrict__ x, const float* __restrict__ wgt,
                              const float* __restrict__ bias, float* __restrict__ y) {
  __shared__ __align__(16) float tile[H_ * W_];
  int bc = blockIdx.x;
  const float* src = x + (size_t)bc * (H_ * W_);
  float* dst = y + (size_t)bc * (H_ * W_);
  int tid = threadIdx.x;

  uint32_t lds0 = (uint32_t)(uintptr_t)&tile[0];
  for (int i = tid; i < (H_ * W_) / 4; i += 256) {       // 576 x 16B DMA
    uint32_t dsa = lds0 + (uint32_t)i * 16u;
    uint64_t ga  = (uint64_t)(uintptr_t)(src + i * 4);
    asm volatile("global_load_async_to_lds_b128 %0, %1, off"
                 :: "v"(dsa), "v"(ga) : "memory");
  }
  // overlap: per-plane dynamic kernel + bias while the DMA is in flight
  float wk[9];
#pragma unroll
  for (int j = 0; j < 9; ++j) wk[j] = wgt[(size_t)bc * 9 + j];
  float bv = bias[bc];
  asm volatile("s_wait_asynccnt 0x0" ::: "memory");      // own wave's DMA done
  __syncthreads();                                       // all waves' data visible

  for (int i = tid; i < H_ * W_; i += 256) {
    int h = i / W_, w = i % W_;
    float acc = bv;
#pragma unroll
    for (int kh = 0; kh < 3; ++kh) {
      int hh = h + kh - 1;
      if ((unsigned)hh >= (unsigned)H_) continue;
#pragma unroll
      for (int kw = 0; kw < 3; ++kw) {
        int ww = w + kw - 1;
        if ((unsigned)ww >= (unsigned)W_) continue;
        acc += tile[hh * W_ + ww] * wk[kh * 3 + kw];
      }
    }
    dst[i] = acc;
  }
}

// ---------------------------------------------------------------------------
extern "C" void kernel_launch(void* const* d_in, const int* in_sizes, int n_in,
                              void* d_out, int out_size, void* d_ws, size_t ws_size,
                              hipStream_t stream) {
  const float* x     = (const float*)d_in[0];
  const float* dw_w  = (const float*)d_in[1];
  const float* dw_b  = (const float*)d_in[2];
  const float* p_w1  = (const float*)d_in[3];
  const float* p_bng = (const float*)d_in[4];
  const float* p_bnb = (const float*)d_in[5];
  const float* p_bnm = (const float*)d_in[6];
  const float* p_bnv = (const float*)d_in[7];
  const float* p_w2  = (const float*)d_in[8];
  const float* p_b2  = (const float*)d_in[9];
  const float* q_w1  = (const float*)d_in[10];
  const float* q_bng = (const float*)d_in[11];
  const float* q_bnb = (const float*)d_in[12];
  const float* q_bnm = (const float*)d_in[13];
  const float* q_bnv = (const float*)d_in[14];
  const float* q_w2  = (const float*)d_in[15];
  const float* q_b2  = (const float*)d_in[16];
  float* y = (float*)d_out;

  // workspace layout (all offsets 256B-aligned)
  char* ws = (char*)d_ws;
  float*    x1t  = (float*)(ws + 0);              // 288*512  f32 = 589824 B
  float*    gapt = (float*)(ws + 589824);         // 32*512   f32 =  65536 B
  _Float16* Hp   = (_Float16*)(ws + 655360);      // 288*256  f16 = 147456 B
  _Float16* Hq   = (_Float16*)(ws + 802816);      // 32*256   f16 =  16384 B
  float*    Sq   = (float*)(ws + 819200);         // 32*1024  f32 = 131072 B
  float*    wout = (float*)(ws + 950272);         // 16384*9  f32 = 589824 B
  float*    bout = (float*)(ws + 1540096);        // 16384    f32 =  65536 B

  // pass 1 over x: pooled stats
  pool_kernel<<<B_ * C_, 288, 0, stream>>>(x, x1t);
  gap_kernel<<<(B_ * C_ + 255) / 256, 256, 0, stream>>>(x1t, gapt);

  // p branch: (288x512)x(512x256) -> BN/GELU -> (288x256)x(256x512) -> softmax mix
  gemm_bn_gelu_kernel<<<dim3(288 / 16, CR_ / 16), 32, 0, stream>>>(
      x1t, p_w1, p_bng, p_bnb, p_bnm, p_bnv, Hp, CR_, C_);
  gemm_p2_fuse_kernel<<<dim3(288 / 16, C_ / 16), 32, 0, stream>>>(
      Hp, p_w2, p_b2, x1t, dw_w, wout);

  // q branch: (32x512)x(512x256) -> BN/GELU -> (32x256)x(256x1024) -> softmax mix
  gemm_bn_gelu_kernel<<<dim3(B_ / 16, CR_ / 16), 32, 0, stream>>>(
      gapt, q_w1, q_bng, q_bnb, q_bnm, q_bnv, Hq, CR_, C_);
  gemm_bias_kernel<<<dim3(B_ / 16, (G_ * C_) / 16), 32, 0, stream>>>(
      Hq, q_w2, q_b2, Sq, G_ * C_, CR_);
  bias_mix_kernel<<<(B_ * C_ + 255) / 256, 256, 0, stream>>>(Sq, dw_b, bout);

  // pass 2 over x (L2-resident): depthwise conv with dynamic kernels
  dwconv_kernel<<<B_ * C_, 256, 0, stream>>>(x, wout, bout, y);
}
